// BodyConvClothGraphConvolution_38405597561665
// MI455X (gfx1250) — compile-verified, matrix operand order checked by reading, not run on hardware
//
#include <hip/hip_runtime.h>
#include <hip/hip_bf16.h>
#include <stdint.h>

// Problem dims (fixed by reference)
#define R_DIM 8192
#define C_DIM 8192
#define D_DIM 256
#define H_DIM 256

typedef __bf16 v16bf __attribute__((ext_vector_type(16)));
typedef float  v8f   __attribute__((ext_vector_type(8)));
typedef float  v4f   __attribute__((ext_vector_type(4)));

union FragU { uint4 q[2]; v16bf v; };   // 32 bytes = 16 bf16 = one A/B fragment

// Pack two fp32 -> two bf16 (truncate) in ONE v_perm_b32:
// result = { hi[31:16], lo[31:16] }
__device__ __forceinline__ uint32_t pkbf(float lo, float hi) {
  return __builtin_amdgcn_perm(__float_as_uint(hi), __float_as_uint(lo), 0x07060302u);
}

__device__ __forceinline__ v8f zero8() {
  v8f z;
#pragma unroll
  for (int i = 0; i < 8; ++i) z[i] = 0.0f;
  return z;
}

// 8 WMMAs on a 32(M) x 64(N) wave tile
__device__ __forceinline__ void wmma8(const FragU a[2], const FragU b[4], v8f acc[2][4]) {
#pragma unroll
  for (int nf = 0; nf < 4; ++nf)
#pragma unroll
    for (int mf = 0; mf < 2; ++mf)
      acc[mf][nf] = __builtin_amdgcn_wmma_f32_16x16x32_bf16(
          false, a[mf].v, false, b[nf].v, (short)0, acc[mf][nf], false, false);
}

// ---------------------------------------------------------------------------
// wT[n][k] (bf16 packed u32) = w[k][n].  256x256, tiny.
__global__ void prep_w_kernel(const float* __restrict__ w, uint32_t* __restrict__ wT) {
  int t = blockIdx.x * blockDim.x + threadIdx.x;   // 0 .. 32767
  int n  = t >> 7;
  int k2 = (t & 127) * 2;
  wT[t] = pkbf(w[(size_t)k2 * H_DIM + n], w[(size_t)(k2 + 1) * H_DIM + n]);
}

// ---------------------------------------------------------------------------
// out rows [R, R+C) = relu(b[n])
__global__ void fill_body_kernel(const float* __restrict__ b, float* __restrict__ out) {
  int t = blockIdx.x * blockDim.x + threadIdx.x;
  float v = b[t & (H_DIM - 1)];
  out[(size_t)R_DIM * H_DIM + t] = v > 0.0f ? v : 0.0f;
}

// ---------------------------------------------------------------------------
// xbT[n][k] (bf16 packed u32) = X0_body[k][n];  X0_body is 8192x256 fp32.
__global__ void transpose_body_kernel(const float* __restrict__ xb,
                                      uint32_t* __restrict__ xbT) {
  __shared__ float tile[32][33];
  int r0 = blockIdx.y * 32;   // k base
  int c0 = blockIdx.x * 32;   // n base
  int tx = threadIdx.x, ty = threadIdx.y;          // (32, 8)
#pragma unroll
  for (int i = 0; i < 32; i += 8)
    tile[ty + i][tx] = xb[(size_t)(r0 + ty + i) * H_DIM + c0 + tx];
  __syncthreads();
  int t = ty * 32 + tx;
#pragma unroll
  for (int p = 0; p < 2; ++p) {
    int nl = p * 16 + (t >> 4);
    int kp = t & 15;
    xbT[((size_t)(c0 + nl) * C_DIM + r0) / 2 + kp] =
        pkbf(tile[kp * 2][nl], tile[kp * 2 + 1][nl]);
  }
}

// ---------------------------------------------------------------------------
// GEMM1: X0 = notes @ w  (16384x256 * 256x256), bf16 WMMA, fully unrolled K.
__global__ void __launch_bounds__(256)
gemm1_kernel(const float* __restrict__ notes, const uint32_t* __restrict__ wT,
             float* __restrict__ out) {
  const int lane  = threadIdx.x & 31;
  const int wid   = threadIdx.x >> 5;
  const int waveM = wid & 3;
  const int waveN = wid >> 2;
  const int mBase = blockIdx.y * 128 + waveM * 32;
  const int nBase = blockIdx.x * 128 + waveN * 64;
  const int lrow  = lane & 15;
  const int lhalf = lane >> 4;

  // one base pointer each; all fragment loads use immediate offsets
  const char* ab = (const char*)notes + ((size_t)(mBase + lrow) * D_DIM + lhalf * 8) * 4;
  const char* bb = (const char*)wT    + ((size_t)(nBase + lrow) * D_DIM + lhalf * 8) * 2;

  v8f acc[2][4];
#pragma unroll
  for (int i = 0; i < 2; ++i)
#pragma unroll
    for (int j = 0; j < 4; ++j) acc[i][j] = zero8();

#pragma unroll
  for (int kt = 0; kt < D_DIM / 32; ++kt) {
    FragU a[2];
#pragma unroll
    for (int mf = 0; mf < 2; ++mf) {
      const char* p = ab + mf * (16 * D_DIM * 4) + kt * 128;
      v4f f0 = *(const v4f*)(p);
      v4f f1 = *(const v4f*)(p + 16);
      v4f g0 = *(const v4f*)(p + 64);
      v4f g1 = *(const v4f*)(p + 80);
      a[mf].q[0] = make_uint4(pkbf(f0.x, f0.y), pkbf(f0.z, f0.w),
                              pkbf(f1.x, f1.y), pkbf(f1.z, f1.w));
      a[mf].q[1] = make_uint4(pkbf(g0.x, g0.y), pkbf(g0.z, g0.w),
                              pkbf(g1.x, g1.y), pkbf(g1.z, g1.w));
    }
    FragU bfr[4];
#pragma unroll
    for (int nf = 0; nf < 4; ++nf) {
      const char* p = bb + nf * (16 * D_DIM * 2) + kt * 64;
      bfr[nf].q[0] = *(const uint4*)(p);
      bfr[nf].q[1] = *(const uint4*)(p + 32);
    }
    wmma8(a, bfr, acc);
  }

  // top blocks (blockIdx.y < 64) -> rows [0,R); body -> rows [R+C, R+2C)
  const size_t rowShift = (blockIdx.y < (R_DIM / 128)) ? 0 : (size_t)C_DIM;
#pragma unroll
  for (int mf = 0; mf < 2; ++mf)
#pragma unroll
    for (int r = 0; r < 8; ++r) {
      size_t m = (size_t)(mBase + mf * 16 + r + lhalf * 8) + rowShift;
#pragma unroll
      for (int nf = 0; nf < 4; ++nf)
        out[m * H_DIM + nBase + nf * 16 + lrow] = acc[mf][nf][r];
    }
}

// ---------------------------------------------------------------------------
// GEMM2: out[:R] = relu(X0_top + weight @ X0_body + b)
// 128x128 block tile, 8 waves * (2x4) wmma frags, K step 32.
// weight: fp32 HBM stream (non-temporal) -> v_perm cvt -> LDS (80B pitch,
// conflict-free b128 frag loads), double buffered; B frags register-resident,
// double buffered; K loop unrolled by 2 (even/odd buffers), 1 barrier/step.
#define BM2 128
#define LDS_PITCH 80
#define BUF_BYTES (BM2 * LDS_PITCH)   // 10240

__device__ __forceinline__ void loadW_at(const float* wp, int koff, v4f s[4]) {
  const char* p = (const char*)wp + koff;
#pragma unroll
  for (int i = 0; i < 4; ++i)
    s[i] = __builtin_nontemporal_load((const v4f*)(p + i * 16));
}

__device__ __forceinline__ void loadB_at(const char* bb, int koff, FragU b[4]) {
#pragma unroll
  for (int nf = 0; nf < 4; ++nf) {
    b[nf].q[0] = *(const uint4*)(bb + nf * (16 * C_DIM * 2) + koff);
    b[nf].q[1] = *(const uint4*)(bb + nf * (16 * C_DIM * 2) + koff + 32);
  }
}

__device__ __forceinline__ void loadA_at(const char* la, int bufoff, FragU a[2]) {
#pragma unroll
  for (int mf = 0; mf < 2; ++mf) {
    a[mf].q[0] = *(const uint4*)(la + bufoff + mf * (16 * LDS_PITCH));
    a[mf].q[1] = *(const uint4*)(la + bufoff + mf * (16 * LDS_PITCH) + 32);
  }
}

__device__ __forceinline__ void stage_at(char* sb, int bufoff, const v4f s[4]) {
  uint4* dst = (uint4*)(sb + bufoff);
  dst[0] = make_uint4(pkbf(s[0].x, s[0].y), pkbf(s[0].z, s[0].w),
                      pkbf(s[1].x, s[1].y), pkbf(s[1].z, s[1].w));
  dst[1] = make_uint4(pkbf(s[2].x, s[2].y), pkbf(s[2].z, s[2].w),
                      pkbf(s[3].x, s[3].y), pkbf(s[3].z, s[3].w));
}

__global__ void __launch_bounds__(256)
gemm2_kernel(const float* __restrict__ weight, const uint32_t* __restrict__ xbT,
             const float* __restrict__ bias, float* __restrict__ out) {
  __shared__ char lds[2 * BUF_BYTES];
  const int tid   = threadIdx.x;
  const int lane  = tid & 31, wid = tid >> 5;
  const int waveM = wid & 3, waveN = wid >> 2;
  const int mBlock = blockIdx.y * BM2;
  const int nBase  = blockIdx.x * 128 + waveN * 64;
  const int lrow   = lane & 15;
  const int lhalf  = lane >> 4;

  // staging: thread -> (row, 16-float half-row) of the 128x32 fp32 weight tile
  const int sRow  = tid >> 1;
  const int sHalf = tid & 1;

  const float* wp = weight + (size_t)(mBlock + sRow) * (size_t)C_DIM + sHalf * 16;
  const char*  bb = (const char*)xbT + ((size_t)(nBase + lrow) * (size_t)C_DIM + lhalf * 8) * 2;
  char*        sb = lds + sRow * LDS_PITCH + sHalf * 32;
  const char*  la = lds + (waveM * 32 + lrow) * LDS_PITCH + lhalf * 16;

  v8f acc[2][4];
#pragma unroll
  for (int i = 0; i < 2; ++i)
#pragma unroll
    for (int j = 0; j < 4; ++j) acc[i][j] = zero8();

  v4f   s[4];
  FragU b0[4], b1[4], a[2];

  loadW_at(wp, 0, s);
  stage_at(sb, 0, s);
  loadB_at(bb, 0, b0);
  __syncthreads();

  const int NT = C_DIM / 32;   // 256 K-steps
  for (int kt = 0; kt < NT - 2; kt += 2) {
    // even step: compute tile kt (buf0, b0); prefetch kt+1
    loadW_at(wp, 128, s);            // fp32 tile kt+1
    loadB_at(bb, 64, b1);            // B frags kt+1
    loadA_at(la, 0, a);
    wmma8(a, b0, acc);
    stage_at(sb, BUF_BYTES, s);      // -> buf1
    __syncthreads();

    // odd step: compute tile kt+1 (buf1, b1); prefetch kt+2
    loadW_at(wp, 256, s);            // fp32 tile kt+2
    loadB_at(bb, 128, b0);           // B frags kt+2
    loadA_at(la, BUF_BYTES, a);
    wmma8(a, b1, acc);
    stage_at(sb, 0, s);              // -> buf0
    __syncthreads();

    wp += 64;                        // 2 K-steps * 32 floats
    bb += 128;                       // 2 K-steps * 64 bytes
  }
  // tail pair: kt = NT-2
  loadW_at(wp, 128, s);
  loadB_at(bb, 64, b1);
  loadA_at(la, 0, a);
  wmma8(a, b0, acc);
  stage_at(sb, BUF_BYTES, s);
  __syncthreads();
  loadA_at(la, BUF_BYTES, a);
  wmma8(a, b1, acc);

  // epilogue: += X0_top (already in out), += bias, relu
  float bn[4];
#pragma unroll
  for (int nf = 0; nf < 4; ++nf) bn[nf] = bias[nBase + nf * 16 + lrow];
#pragma unroll
  for (int mf = 0; mf < 2; ++mf)
#pragma unroll
    for (int r = 0; r < 8; ++r) {
      int m = mBlock + waveM * 32 + mf * 16 + r + lhalf * 8;
#pragma unroll
      for (int nf = 0; nf < 4; ++nf) {
        size_t idx = (size_t)m * H_DIM + nBase + nf * 16 + lrow;
        float v = acc[mf][nf][r] + out[idx] + bn[nf];
        out[idx] = v > 0.0f ? v : 0.0f;
      }
    }
}

// ---------------------------------------------------------------------------
extern "C" void kernel_launch(void* const* d_in, const int* in_sizes, int n_in,
                              void* d_out, int out_size, void* d_ws, size_t ws_size,
                              hipStream_t stream) {
  (void)in_sizes; (void)n_in; (void)out_size; (void)ws_size;
  const float* notes  = (const float*)d_in[0];   // (R+C, D)
  const float* weight = (const float*)d_in[1];   // (R, C)
  const float* w      = (const float*)d_in[2];   // (D, H)
  const float* b      = (const float*)d_in[3];   // (H,)
  float* out = (float*)d_out;                    // (R+2C, H)

  uint32_t* wT  = (uint32_t*)d_ws;                              // 128 KB: w^T bf16
  uint32_t* xbT = (uint32_t*)((char*)d_ws + (128 << 10));       // 4 MB: X0_body^T bf16

  prep_w_kernel<<<128, 256, 0, stream>>>(w, wT);
  fill_body_kernel<<<(C_DIM * H_DIM) / 256, 256, 0, stream>>>(b, out);
  gemm1_kernel<<<dim3(2, 128), 256, 0, stream>>>(notes, wT, out);
  transpose_body_kernel<<<dim3(8, 256), dim3(32, 8), 0, stream>>>(
      out + (size_t)(R_DIM + C_DIM) * H_DIM, xbT);
  gemm2_kernel<<<dim3(2, 64), 256, 0, stream>>>(weight, xbT, b, out);
}
